// BasicBlock_52201032516127
// MI455X (gfx1250) — matrix operand order repaired and checked by reading, not compiled, not run on hardware
//
#include <hip/hip_runtime.h>
#include <hip/hip_bf16.h>

typedef __attribute__((ext_vector_type(16))) _Float16 v16h;
typedef __attribute__((ext_vector_type(8)))  float    v8f;
typedef __attribute__((ext_vector_type(2)))  _Float16 h2vec;

#define BATCH 16
#define CH    64
#define HT    128
#define WD    128
#define HWSZ  (HT * WD)           // 16384
#define MIP   8
#define LDS_COLS 34
#define CI_STRIDE 66              // padded channel stride in LDS (halfs)

// ---------------------------------------------------------------------------
// Weight pre-swizzle: OIHW fp32 -> f16 WMMA A-fragments (16x16x32 lane layout)
// index: ((conv*36 + co_block*9 + tap)*2 + kc)*512 + lane*16 + e
// ---------------------------------------------------------------------------
__global__ void prep_wfrag(const float* __restrict__ w1,
                           const float* __restrict__ w2,
                           _Float16* __restrict__ wfrag) {
    int gid      = blockIdx.x;             // 144 groups
    int kc       = gid & 1;
    int tap      = (gid >> 1) % 9;
    int co_block = (gid / 18) & 3;
    int conv     = gid / 72;
    const float* w = conv ? w2 : w1;
    int t    = threadIdx.x;                // 0..511
    int lane = t >> 4;
    int e    = t & 15;
    int m  = lane & 15, lh = lane >> 4;
    int vv = e >> 1,    kp = e & 1;
    // A-fragment K order used here (B at line 168 matches): contiguous K, lane-half h holds K 16h..16h+15;
    // each VGPR holds a K pair.
    int K0 = lh * 16 + vv * 2 + kp;
    int ci = kc * 32 + K0;
    int co = co_block * 16 + m;
    int kh = tap / 3, kw = tap % 3;
    float val = w[((co * CH + ci) * 3 + kh) * 3 + kw];
    wfrag[(((conv * 36 + co_block * 9 + tap) * 2 + kc) * 32 + lane) * 16 + e] =
        (_Float16)val;
}

// ---------------------------------------------------------------------------
// Mean over H*W per (b,c)
// ---------------------------------------------------------------------------
__global__ void mean_hw(const float* __restrict__ src, float* __restrict__ out) {
    __shared__ float red[256];
    int bc = blockIdx.x;                   // 0..1023
    const float* p = src + (size_t)bc * HWSZ;
    float s = 0.f;
    for (int i = threadIdx.x; i < HWSZ; i += 256) s += p[i];
    red[threadIdx.x] = s;
    __syncthreads();
    for (int off = 128; off > 0; off >>= 1) {
        if (threadIdx.x < off) red[threadIdx.x] += red[threadIdx.x + off];
        __syncthreads();
    }
    if (threadIdx.x == 0) out[bc] = red[0] * (1.f / (float)HWSZ);
}

// c[b,co] = (mean[b] . wcm[co] + bcm[co] > 0) ? 1 : 0
__global__ void channel_mask(const float* __restrict__ mean,
                             const float* __restrict__ wcm,
                             const float* __restrict__ bcm,
                             float* __restrict__ out) {
    int i  = blockIdx.x * 256 + threadIdx.x;   // 0..1023
    int b  = i >> 6, co = i & 63;
    float s = bcm[co];
    const float* m = mean + b * CH;
    const float* w = wcm + co * CH;
    for (int c = 0; c < CH; ++c) s += m[c] * w[c];
    out[i] = (s > 0.f) ? 1.f : 0.f;
}

// s_hard[b,h,w] = (sum_c x*w_sm + b_sm) > 0
__global__ void spatial_mask(const float* __restrict__ x,
                             const float* __restrict__ wsm,
                             const float* __restrict__ bsm,
                             float* __restrict__ sh) {
    int p  = blockIdx.x * 256 + threadIdx.x;   // < 262144
    int b  = p >> 14;
    int hw = p & (HWSZ - 1);
    float s = bsm[0];
    const float* px = x + (size_t)b * CH * HWSZ + hw;
    for (int c = 0; c < CH; ++c) s += px[c * HWSZ] * wsm[c];
    sh[p] = (s > 0.f) ? 1.f : 0.f;
}

// 3x3 max dilation (zero outside = skip; mask values are {0,1})
__global__ void dilate3x3(const float* __restrict__ sh, float* __restrict__ sd) {
    int p  = blockIdx.x * 256 + threadIdx.x;
    int b  = p >> 14;
    int hw = p & (HWSZ - 1);
    int h  = hw >> 7, w = hw & 127;
    float m = 0.f;
    for (int dh = -1; dh <= 1; ++dh)
        for (int dw = -1; dw <= 1; ++dw) {
            int hh = h + dh, ww = w + dw;
            if ((unsigned)hh < (unsigned)HT && (unsigned)ww < (unsigned)WD)
                m = fmaxf(m, sh[(b << 14) + (hh << 7) + ww]);
        }
    sd[p] = m;
}

// ---------------------------------------------------------------------------
// Implicit-GEMM 3x3 conv via v_wmma_f32_16x16x32_f16.
// Block: 256 thr = 8 waves. Tile: 64 Cout x 32 pixels of one output row.
// Gating: out = act(conv+bias) * cmask[b,co] * smask[b,h,w]
// ---------------------------------------------------------------------------
__global__ __launch_bounds__(256) void conv3x3_wmma(
    const float*    __restrict__ src,     // (B,64,H,W) fp32
    const _Float16* __restrict__ wfrag,   // pre-swizzled fragments for this conv
    const float*    __restrict__ bias,    // (64)
    const float*    __restrict__ cmask,   // (B,64)
    const float*    __restrict__ smask,   // (B,H,W)
    float*          __restrict__ dst,     // (B,64,H,W)
    int do_relu) {
    __shared__ _Float16 lds[3 * LDS_COLS * CI_STRIDE];

    const int tid  = threadIdx.x;
    const int bid  = blockIdx.x;          // B*H*(W/32) = 8192
    const int wblk = bid & 3;
    const int h    = (bid >> 2) & (HT - 1);
    const int b    = bid >> 9;
    const int w0   = wblk * 32;

    // hint: keep this block's weight fragments hot
    __builtin_prefetch(wfrag + ((tid >> 5) & 3) * 9 * 2 * 512, 0, 0);

    // Stage 64ci x 3row x 34col halo into LDS as f16, [row][col][ci] padded.
    for (int idx = tid; idx < CH * 3 * LDS_COLS; idx += 256) {
        int ci  = idx / (3 * LDS_COLS);
        int r2  = idx - ci * (3 * LDS_COLS);
        int row = r2 / LDS_COLS;
        int col = r2 - row * LDS_COLS;
        int hin = h + row - 1;
        int win = w0 + col - 1;
        float v = 0.f;
        if ((unsigned)hin < (unsigned)HT && (unsigned)win < (unsigned)WD)
            v = src[((size_t)(b * CH + ci) * HT + hin) * WD + win];
        lds[(row * LDS_COLS + col) * CI_STRIDE + ci] = (_Float16)v;
    }
    __syncthreads();

    const int wv       = tid >> 5;        // wave 0..7
    const int lane     = tid & 31;
    const int co_block = wv & 3;          // 4 blocks of 16 Cout
    const int pb       = wv >> 2;         // 2 blocks of 16 pixels
    const int lh       = lane >> 4;       // lane half
    const int n        = lane & 15;

    v8f acc = {};
    #pragma unroll
    for (int tap = 0; tap < 9; ++tap) {
        const int kh = tap / 3, kw = tap - kh * 3;
        #pragma unroll
        for (int kc = 0; kc < 2; ++kc) {
            // A fragment: one aligned 32B load per lane (pre-swizzled layout)
            v16h a = *(const v16h*)(wfrag +
                (((co_block * 9 + tap) * 2 + kc) * 32 + lane) * 16);
            // B fragment (32x16): lane half selects K 0..15 / 16..31, n = column
            v16h bm;
            const int colb = pb * 16 + n + kw;
            const _Float16* lp =
                &lds[(kh * LDS_COLS + colb) * CI_STRIDE + kc * 32 + lh * 16];
            #pragma unroll
            for (int q = 0; q < 8; ++q) {
                h2vec pr = *(const h2vec*)(lp + q * 2);
                bm[2 * q]     = pr.x;
                bm[2 * q + 1] = pr.y;
            }
            acc = __builtin_amdgcn_wmma_f32_16x16x32_f16(
                false, a, false, bm, (short)0, acc, false, false);
        }
    }

    // Epilogue: bias, activation, channel & spatial gating.
    const int   wpix = w0 + pb * 16 + n;
    const float sm   = smask[(b * HT + h) * WD + wpix];
    #pragma unroll
    for (int r = 0; r < 8; ++r) {
        int   co = co_block * 16 + lh * 8 + r;    // C/D layout: VGPR r -> M = lh*8+r
        float v  = acc[r] + bias[co];
        if (do_relu) v = fmaxf(v, 0.f);
        v *= cmask[b * CH + co] * sm;
        dst[((size_t)(b * CH + co) * HT + h) * WD + wpix] = v;
    }
}

// ---------------------------------------------------------------------------
// CoordAtt helpers
// ---------------------------------------------------------------------------
__global__ void pool_h(const float* __restrict__ z, float* __restrict__ xh) {
    int i = blockIdx.x * 256 + threadIdx.x;        // (b*64+c)*128 + h
    const float* p = z + (size_t)i * WD;
    float s = 0.f;
    for (int w = 0; w < WD; ++w) s += p[w];
    xh[i] = s * (1.f / (float)WD);
}

__global__ void pool_w(const float* __restrict__ z, float* __restrict__ xw) {
    int i  = blockIdx.x * 256 + threadIdx.x;       // (b*64+c)*128 + w
    int bc = i >> 7, w = i & 127;
    const float* p = z + (size_t)bc * HWSZ + w;
    float s = 0.f;
    for (int hh = 0; hh < HT; ++hh) s += p[hh << 7];
    xw[i] = s * (1.f / (float)HT);
}

__global__ void coord_mid(const float* __restrict__ xh, const float* __restrict__ xw,
                          const float* __restrict__ w1, const float* __restrict__ b1,
                          const float* __restrict__ gamma, const float* __restrict__ beta,
                          float* __restrict__ t) {
    int i   = blockIdx.x * 256 + threadIdx.x;      // < 16*8*256
    int pos = i & 255;
    int mip = (i >> 8) & (MIP - 1);
    int b   = i >> 11;
    float s = b1[mip];
    const float* w = w1 + mip * CH;
    const float* src = (pos < HT) ? (xh + (b * CH) * HT + pos)
                                  : (xw + (b * CH) * WD + (pos - HT));
    for (int c = 0; c < CH; ++c) s += src[c * HT] * w[c];
    s = s * gamma[mip] + beta[mip];
    t[i] = s * fminf(fmaxf(s + 3.f, 0.f), 6.f) * (1.f / 6.f);   // hswish
}

__global__ void coord_att(const float* __restrict__ t,
                          const float* __restrict__ wh, const float* __restrict__ bh,
                          const float* __restrict__ ww, const float* __restrict__ bw,
                          float* __restrict__ ah, float* __restrict__ aw) {
    int i    = blockIdx.x * 256 + threadIdx.x;     // < 262144
    int part = i >> 17;                            // 0: a_h, 1: a_w
    int j    = i & 131071;
    int pos  = j & 127;
    int c    = (j >> 7) & 63;
    int b    = j >> 13;
    const float* wsel = part ? ww : wh;
    const float* bsel = part ? bw : bh;
    int tpos = part ? (HT + pos) : pos;
    float s = bsel[c];
    const float* tp = t + (b * MIP) * 256 + tpos;
    const float* wv = wsel + c * MIP;
    for (int m = 0; m < MIP; ++m) s += tp[m * 256] * wv[m];
    (part ? aw : ah)[j] = 1.f / (1.f + __expf(-s));
}

// out = relu(z * a_h * a_w + x), in place on d_out (z lives there)
__global__ void fuse_out(const float* __restrict__ x,
                         const float* __restrict__ ah, const float* __restrict__ aw,
                         float* __restrict__ out) {
    size_t i  = (size_t)blockIdx.x * 256 + threadIdx.x;
    int    w  = (int)(i & 127);
    size_t r  = i >> 7;
    int    hh = (int)(r & 127);
    size_t bc = r >> 7;
    float  z  = out[i];
    float  v  = z * ah[bc * HT + hh] * aw[bc * WD + w] + x[i];
    out[i] = fmaxf(v, 0.f);
}

// ---------------------------------------------------------------------------
extern "C" void kernel_launch(void* const* d_in, const int* in_sizes, int n_in,
                              void* d_out, int out_size, void* d_ws, size_t ws_size,
                              hipStream_t stream) {
    const float* x      = (const float*)d_in[0];
    const float* w_sm   = (const float*)d_in[1];
    const float* b_sm   = (const float*)d_in[2];
    const float* w_cm1  = (const float*)d_in[3];
    const float* b_cm1  = (const float*)d_in[4];
    const float* w_cm2  = (const float*)d_in[5];
    const float* b_cm2  = (const float*)d_in[6];
    const float* w1     = (const float*)d_in[7];
    const float* b1     = (const float*)d_in[8];
    const float* w2     = (const float*)d_in[9];
    const float* b2     = (const float*)d_in[10];
    const float* ca_w1  = (const float*)d_in[11];
    const float* ca_b1  = (const float*)d_in[12];
    const float* ca_g   = (const float*)d_in[13];
    const float* ca_bt  = (const float*)d_in[14];
    const float* ca_wh  = (const float*)d_in[15];
    const float* ca_bh  = (const float*)d_in[16];
    const float* ca_ww  = (const float*)d_in[17];
    const float* ca_bw  = (const float*)d_in[18];
    float* out = (float*)d_out;

    // workspace layout (floats)
    float* ws    = (float*)d_ws;
    float* y     = ws;                               // 16777216
    float* sh    = y     + (size_t)BATCH * CH * HWSZ;
    float* sd    = sh    + BATCH * HWSZ;             // 262144 each
    float* meanx = sd    + BATCH * HWSZ;             // 1024
    float* c1m   = meanx + 1024;
    float* meany = c1m   + 1024;
    float* c2m   = meany + 1024;
    float* xh    = c2m   + 1024;                     // 131072
    float* xw    = xh    + BATCH * CH * HT;          // 131072
    float* tmid  = xw    + BATCH * CH * WD;          // 32768
    float* ah    = tmid  + BATCH * MIP * 256;        // 131072
    float* aw    = ah    + BATCH * CH * HT;          // 131072
    _Float16* wfrag = (_Float16*)(aw + BATCH * CH * WD); // 73728 halfs

    // 1. pre-swizzle conv weights into WMMA fragment layout
    prep_wfrag<<<144, 512, 0, stream>>>(w1, w2, wfrag);
    // 2. channel mask 1 from input means
    mean_hw<<<BATCH * CH, 256, 0, stream>>>(x, meanx);
    channel_mask<<<4, 256, 0, stream>>>(meanx, w_cm1, b_cm1, c1m);
    // 3. spatial hard mask + dilation
    spatial_mask<<<BATCH * HWSZ / 256, 256, 0, stream>>>(x, w_sm, b_sm, sh);
    dilate3x3<<<BATCH * HWSZ / 256, 256, 0, stream>>>(sh, sd);
    // 4. conv1 (WMMA): relu, gated by c1 & dilated mask -> y
    conv3x3_wmma<<<BATCH * HT * (WD / 32), 256, 0, stream>>>(
        x, wfrag, b1, c1m, sd, y, 1);
    // 5. channel mask 2 from y means
    mean_hw<<<BATCH * CH, 256, 0, stream>>>(y, meany);
    channel_mask<<<4, 256, 0, stream>>>(meany, w_cm2, b_cm2, c2m);
    // 6. conv2 (WMMA): no act, gated by c2 & hard mask -> z (in d_out)
    conv3x3_wmma<<<BATCH * HT * (WD / 32), 256, 0, stream>>>(
        y, wfrag + 36864, b2, c2m, sh, out, 0);
    // 7. CoordAtt
    pool_h<<<BATCH * CH * HT / 256, 256, 0, stream>>>(out, xh);
    pool_w<<<BATCH * CH * WD / 256, 256, 0, stream>>>(out, xw);
    coord_mid<<<BATCH * MIP * 256 / 256, 256, 0, stream>>>(
        xh, xw, ca_w1, ca_b1, ca_g, ca_bt, tmid);
    coord_att<<<2 * BATCH * CH * HT / 256, 256, 0, stream>>>(
        tmid, ca_wh, ca_bh, ca_ww, ca_bw, ah, aw);
    // 8. out = relu(z * a_h * a_w + x)
    fuse_out<<<BATCH * CH * HWSZ / 256, 256, 0, stream>>>(x, ah, aw, out);
}
